// AttnFFN_12987981103356
// MI455X (gfx1250) — compile-verified, matrix operand order checked by reading, not run on hardware
//
#include <hip/hip_runtime.h>
#include <hip/hip_bf16.h>

typedef __attribute__((ext_vector_type(16))) _Float16 v16h;
typedef __attribute__((ext_vector_type(8)))  float    v8f;

#define WMMA_F16(a,b,c) __builtin_amdgcn_wmma_f32_16x16x32_f16(false,(a),false,(b),(short)0,(c),false,false)

// Problem constants
#define BSZ    256
#define BDIM   384
#define RRES   7
#define NTOK   49
#define HEADS  8
#define KDIM   32
#define DHEAD  128
#define DHTOT  1024
#define NHKD   256
#define HIDDEN 1536
#define NBIAS  49

__device__ __forceinline__ v8f zero8() {
    v8f z;
#pragma unroll
    for (int i = 0; i < 8; ++i) z[i] = 0.f;
    return z;
}

__device__ __forceinline__ v16h zero16h() {
    v16h z;
#pragma unroll
    for (int i = 0; i < 16; ++i) z[i] = (_Float16)0.f;
    return z;
}

// ---------------------------------------------------------------------------
// Generic channel GEMM: out[b,oc,n] = sum_ic W[oc,ic]*X[b,ic,n] + bias[oc]
// epi: 0 = bias, 1 = bias+relu, 2 = res + ls[oc]*(acc+bias)
// grid = (B, OC/128), block = 256 (8 waves; wave w owns a 16-row OC tile).
// Operands staged in LDS in WMMA fragment order: [half-group][row][16 halves],
// so each fragment is one contiguous 32B LDS read (2 x ds_load_b128).
// ---------------------------------------------------------------------------
__global__ __launch_bounds__(256) void gemm_wmma_kernel(
    const float* __restrict__ W, const float* __restrict__ X,
    const float* __restrict__ bias, const float* __restrict__ res,
    const float* __restrict__ ls, float* __restrict__ out,
    int OC, int IC, int epi)
{
    __shared__ __align__(32) _Float16 asW[2 * 128 * 16];  // 8 KB: W K-chunk, frag order
    __shared__ __align__(32) _Float16 bsX[2 * 64 * 16];   // 4 KB: X K-chunk, frag order

    const int b    = blockIdx.x;
    const int tid  = threadIdx.x;
    const int wave = tid >> 5;
    const int lane = tid & 31;
    const int mrow = lane & 15;
    const int half = lane >> 4;
    const int ocb  = blockIdx.y * 128;
    const int oc_base = ocb + wave * 16;

    const float* Xb = X + (size_t)b * IC * NTOK;

    v8f acc[4];
#pragma unroll
    for (int nt = 0; nt < 4; ++nt) acc[nt] = zero8();

    for (int k0 = 0; k0 < IC; k0 += 32) {
        __syncthreads();
        // ---- stage W[ocb..ocb+127][k0..k0+31]: thread -> (oc_local, hg) segment
        {
            const int oc_local = tid >> 1;
            const int hg       = tid & 1;
            const float* wr = W + (size_t)(ocb + oc_local) * IC + k0;
            _Float16* seg = asW + ((hg * 128) + oc_local) * 16;
#pragma unroll
            for (int e = 0; e < 8; ++e) {
                seg[e]     = (_Float16)wr[hg * 8 + e];
                seg[8 + e] = (_Float16)wr[16 + hg * 8 + e];
            }
        }
        // ---- stage X[k0..k0+31][0..63]: threads 0..127 -> (hg, col) segment
        if (tid < 128) {
            const int col = tid & 63;
            const int hg  = tid >> 6;
            const bool inr = (col < NTOK);
            const float* xc = Xb + (size_t)k0 * NTOK + col;
            _Float16* seg = bsX + ((hg * 64) + col) * 16;
#pragma unroll
            for (int e = 0; e < 8; ++e) {
                seg[e]     = (_Float16)(inr ? xc[(hg * 8 + e) * NTOK] : 0.f);
                seg[8 + e] = (_Float16)(inr ? xc[(16 + hg * 8 + e) * NTOK] : 0.f);
            }
        }
        __syncthreads();

        const v16h af = *(const v16h*)(asW + ((half * 128) + wave * 16 + mrow) * 16);
#pragma unroll
        for (int nt = 0; nt < 4; ++nt) {
            const v16h bf = *(const v16h*)(bsX + ((half * 64) + nt * 16 + mrow) * 16);
            acc[nt] = WMMA_F16(af, bf, acc[nt]);
        }
    }

    // epilogue: C layout -> oc = oc_base + j + 8*half, n = nt*16 + mrow
#pragma unroll
    for (int nt = 0; nt < 4; ++nt) {
        const int n = nt * 16 + mrow;
        if (n >= NTOK) continue;
#pragma unroll
        for (int j = 0; j < 8; ++j) {
            const int oc = oc_base + j + 8 * half;
            float v = acc[nt][j] + bias[oc];
            const size_t idx = ((size_t)b * OC + oc) * NTOK + n;
            if (epi == 1)       v = fmaxf(v, 0.f);
            else if (epi == 2)  v = res[idx] + ls[oc] * v;
            out[idx] = v;
        }
    }
}

// ---------------------------------------------------------------------------
// Depthwise 3x3 conv on 7x7 maps (+bias, optional relu). Tiny FLOPs -> VALU.
// ---------------------------------------------------------------------------
__global__ __launch_bounds__(256) void dwconv3x3_kernel(
    const float* __restrict__ in, const float* __restrict__ w,
    const float* __restrict__ bias, float* __restrict__ out,
    int C, int relu, int total)
{
    int idx = blockIdx.x * 256 + threadIdx.x;
    if (idx >= total) return;
    int n = idx % NTOK;
    int c = (idx / NTOK) % C;
    int y = n / RRES, x = n % RRES;
    const float* ip = in + (size_t)(idx / NTOK) * NTOK;
    const float* wp = w + c * 9;
    float acc = bias[c];
#pragma unroll
    for (int dy = -1; dy <= 1; ++dy)
#pragma unroll
        for (int dx = -1; dx <= 1; ++dx) {
            int yy = y + dy, xx = x + dx;
            if (yy >= 0 && yy < RRES && xx >= 0 && xx < RRES)
                acc += ip[yy * RRES + xx] * wp[(dy + 1) * 3 + (dx + 1)];
        }
    if (relu) acc = fmaxf(acc, 0.f);
    out[idx] = acc;
}

// ---------------------------------------------------------------------------
// Fused attention: per block = one batch image; wave h = head h.
//  A) logits = scale*(q^T k) + pos-bias   (WMMA, fragment-ordered q/k in LDS)
//  B) talking-heads-1 + softmax           (VALU over attn_s)
//  C) talking-heads-2 -> writes f16 attention in WMMA fragment order (attn_h,
//     aliased onto the q/k pool which is dead after phase A)
//  D) o = relu(attn @ v^T + v_local)      (WMMA; B-frags are direct v16h loads)
// ---------------------------------------------------------------------------
#define FRAG_POOL_HALVES (2 * HEADS * 2 * 64 * 16)   // qs + ks = 32768 halves (64 KB)
#define QK_SEGS          (HEADS * 2 * 64)            // 1024 segments of 16 halves each
#define ATTNH_HALVES     (HEADS * 2 * 2 * 64 * 16)   // 32768 halves (64 KB region, aliases pool)

__global__ __launch_bounds__(256) void attn_kernel(
    const float* __restrict__ q, const float* __restrict__ k,
    const float* __restrict__ v, const float* __restrict__ vlocal,
    const float* __restrict__ th1_w, const float* __restrict__ th1_b,
    const float* __restrict__ th2_w, const float* __restrict__ th2_b,
    const float* __restrict__ attn_bias, const int* __restrict__ bias_idxs,
    float* __restrict__ o)
{
    // 64 KB pool: phase A holds qs|ks (32 KB each); phases C/D reuse it as attn_h
    __shared__ __align__(32) _Float16 pool[FRAG_POOL_HALVES];
    __shared__ float attn_s[HEADS * NTOK * NTOK];    // 76.8 KB logits
    __shared__ float th1s[64], th2s[64], th1bs[8], th2bs[8];

    _Float16* qs     = pool;                         // [h][hg][tok][16]
    _Float16* ks     = pool + QK_SEGS * 16;          // [h][hg][tok][16]
    _Float16* attn_h = pool;                         // [h][kc][hg][n][16]

    const int b    = blockIdx.x;
    const int tid  = threadIdx.x;
    const int h    = tid >> 5;
    const int lane = tid & 31;
    const int mrow = lane & 15;
    const int half = lane >> 4;

    if (tid < 64) { th1s[tid] = th1_w[tid]; th2s[tid] = th2_w[tid]; }
    if (tid < 8)  { th1bs[tid] = th1_b[tid]; th2bs[tid] = th2_b[tid]; }

    const float scale = 0.17677669529663687f;   // 32^-0.5
    const float* qb = q + (size_t)b * NHKD * NTOK;
    const float* kb = k + (size_t)b * NHKD * NTOK;

    // ---- stage q/k into fragment-ordered LDS (coalesced across threads)
#pragma unroll
    for (int s = 0; s < 4; ++s) {
        const int seg = tid + s * 256;               // 0..1023
        const int sh  = seg >> 7;                    // head
        const int hg  = (seg >> 6) & 1;              // half group
        const int tok = seg & 63;
        const bool inr = (tok < NTOK);
        const float* qp = qb + (size_t)(sh * KDIM) * NTOK + tok;
        const float* kp = kb + (size_t)(sh * KDIM) * NTOK + tok;
        _Float16* sq = qs + seg * 16;
        _Float16* sk = ks + seg * 16;
#pragma unroll
        for (int e = 0; e < 8; ++e) {
            const int c0 = hg * 8 + e, c1 = 16 + hg * 8 + e;
            sq[e]     = (_Float16)(inr ? qp[c0 * NTOK] : 0.f);
            sq[8 + e] = (_Float16)(inr ? qp[c1 * NTOK] : 0.f);
            sk[e]     = (_Float16)(inr ? kp[c0 * NTOK] : 0.f);
            sk[8 + e] = (_Float16)(inr ? kp[c1 * NTOK] : 0.f);
        }
    }
    __syncthreads();

    // ---- load all fragments to registers (pool is reused after next barrier)
    v16h afr[4], bfr[4];
#pragma unroll
    for (int t = 0; t < 4; ++t) {
        afr[t] = *(const v16h*)(qs + (((h * 2) + half) * 64 + t * 16 + mrow) * 16);
        bfr[t] = *(const v16h*)(ks + (((h * 2) + half) * 64 + t * 16 + mrow) * 16);
    }
    __syncthreads();   // all q/k reads done; pool may now be overwritten

    // ---- zero attn_h padding region (aliases pool) while WMMAs run
    {
        const v16h z = zero16h();
#pragma unroll
        for (int i = 0; i < 8; ++i)
            *(v16h*)(attn_h + (tid * 8 + i) * 16) = z;
    }

    // ---- Phase A: 16 WMMAs per wave
    {
        v8f catt[4][4];
#pragma unroll
        for (int mt = 0; mt < 4; ++mt)
#pragma unroll
            for (int nt = 0; nt < 4; ++nt) catt[mt][nt] = zero8();

#pragma unroll
        for (int mt = 0; mt < 4; ++mt)
#pragma unroll
            for (int nt = 0; nt < 4; ++nt)
                catt[mt][nt] = WMMA_F16(afr[mt], bfr[nt], catt[mt][nt]);

        // epilogue: scale + positional bias -> attn_s
#pragma unroll
        for (int mt = 0; mt < 4; ++mt)
#pragma unroll
            for (int nt = 0; nt < 4; ++nt) {
                const int m = nt * 16 + mrow;
                if (m >= NTOK) continue;
#pragma unroll
                for (int j = 0; j < 8; ++j) {
                    const int n = mt * 16 + j + 8 * half;
                    if (n < NTOK) {
                        float val = catt[mt][nt][j] * scale
                                  + attn_bias[h * NBIAS + bias_idxs[n * NTOK + m]];
                        attn_s[(h * NTOK + n) * NTOK + m] = val;
                    }
                }
            }
    }
    __syncthreads();

    // ---- Phase B: th1 mix + softmax (two rounds split by n; in-place on attn_s)
    for (int s = 0; s < 2; ++s) {
        float logits[NTOK];
        const int r = tid + s * 256;
        const bool live = (r < HEADS * NTOK);
        const int n = r >> 3, i = r & 7;
        if (live) {
            for (int m = 0; m < NTOK; ++m) {
                float acc = th1bs[i];
#pragma unroll
                for (int j = 0; j < 8; ++j)
                    acc += th1s[i * 8 + j] * attn_s[(j * NTOK + n) * NTOK + m];
                logits[m] = acc;
            }
        }
        __syncthreads();
        if (live) {
            float mx = -1e30f;
            for (int m = 0; m < NTOK; ++m) mx = fmaxf(mx, logits[m]);
            float sum = 0.f;
            for (int m = 0; m < NTOK; ++m) { float e = __expf(logits[m] - mx); logits[m] = e; sum += e; }
            float inv = 1.f / sum;
            for (int m = 0; m < NTOK; ++m) attn_s[(i * NTOK + n) * NTOK + m] = logits[m] * inv;
        }
        __syncthreads();
    }

    // ---- Phase C: th2 mix, written directly in f16 WMMA fragment order
    for (int s = 0; s < 2; ++s) {
        const int r = tid + s * 256;
        if (r < HEADS * NTOK) {
            const int n = r >> 3, i = r & 7;
            for (int m = 0; m < NTOK; ++m) {
                float acc = th2bs[i];
#pragma unroll
                for (int j = 0; j < 8; ++j)
                    acc += th2s[i * 8 + j] * attn_s[(j * NTOK + n) * NTOK + m];
                // inverse fragment mapping: m -> (kc, hg, e)
                const int kc = m >> 5;
                const int K  = m & 31;
                const int hg = (K >> 3) & 1;
                const int e  = (K & 7) + ((K >= 16) ? 8 : 0);
                attn_h[((((i * 2) + kc) * 2 + hg) * 64 + n) * 16 + e] = (_Float16)acc;
            }
        }
    }
    __syncthreads();

    // ---- Phase D: o[d,n] = relu( sum_m v[d,m]*attn[n,m] + v_local[d,n] )
    const float* vh  = v      + ((size_t)b * DHTOT + h * DHEAD) * NTOK;
    const float* vlh = vlocal + ((size_t)b * DHTOT + h * DHEAD) * NTOK;
    float*       oh  = o      + ((size_t)b * DHTOT + h * DHEAD) * NTOK;

    for (int dt = 0; dt < 8; ++dt) {
        v8f co[4];
#pragma unroll
        for (int nt = 0; nt < 4; ++nt) co[nt] = zero8();

#pragma unroll
        for (int kc = 0; kc < 2; ++kc) {
            // A(M=d, K=m) = v[d][m]; rows contiguous in memory
            v16h af;
            const int drow = dt * 16 + mrow;
#pragma unroll
            for (int e = 0; e < 8; ++e) {
                const int m0 = kc * 32 + half * 8 + e;
                const int m1 = kc * 32 + 16 + half * 8 + e;
                af[e]     = (_Float16)((m0 < NTOK) ? vh[drow * NTOK + m0] : 0.f);
                af[8 + e] = (_Float16)((m1 < NTOK) ? vh[drow * NTOK + m1] : 0.f);
            }
#pragma unroll
            for (int nt = 0; nt < 4; ++nt) {
                const v16h bf = *(const v16h*)(attn_h +
                    ((((h * 2) + kc) * 2 + half) * 64 + nt * 16 + mrow) * 16);
                co[nt] = WMMA_F16(af, bf, co[nt]);
            }
        }
#pragma unroll
        for (int nt = 0; nt < 4; ++nt) {
            const int n = nt * 16 + mrow;
            if (n >= NTOK) continue;
#pragma unroll
            for (int j = 0; j < 8; ++j) {
                const int d = dt * 16 + j + 8 * half;
                float val = co[nt][j] + vlh[d * NTOK + n];
                oh[d * NTOK + n] = fmaxf(val, 0.f);
            }
        }
    }
}

// ---------------------------------------------------------------------------
extern "C" void kernel_launch(void* const* d_in, const int* in_sizes, int n_in,
                              void* d_out, int out_size, void* d_ws, size_t ws_size,
                              hipStream_t stream) {
    const float* x         = (const float*)d_in[0];
    const float* q_w       = (const float*)d_in[1];
    const float* q_b       = (const float*)d_in[2];
    const float* k_w       = (const float*)d_in[3];
    const float* k_b       = (const float*)d_in[4];
    const float* v_w       = (const float*)d_in[5];
    const float* v_b       = (const float*)d_in[6];
    const float* lv_w      = (const float*)d_in[7];
    const float* lv_b      = (const float*)d_in[8];
    const float* th1_w     = (const float*)d_in[9];
    const float* th1_b     = (const float*)d_in[10];
    const float* th2_w     = (const float*)d_in[11];
    const float* th2_b     = (const float*)d_in[12];
    const float* attn_bias = (const float*)d_in[13];
    const float* proj_w    = (const float*)d_in[14];
    const float* proj_b    = (const float*)d_in[15];
    const float* fc1_w     = (const float*)d_in[16];
    const float* fc1_b     = (const float*)d_in[17];
    const float* mid_w     = (const float*)d_in[18];
    const float* mid_b     = (const float*)d_in[19];
    const float* fc2_w     = (const float*)d_in[20];
    const float* fc2_b     = (const float*)d_in[21];
    const float* ls1       = (const float*)d_in[22];
    const float* ls2       = (const float*)d_in[23];
    const int*   bias_idxs = (const int*)d_in[24];

    // workspace layout (f32): ~88.3M floats (~353 MB)
    const size_t SZ_QK = (size_t)BSZ * NHKD   * NTOK;
    const size_t SZ_V  = (size_t)BSZ * DHTOT  * NTOK;
    const size_t SZ_Y  = (size_t)BSZ * BDIM   * NTOK;
    const size_t SZ_H  = (size_t)BSZ * HIDDEN * NTOK;
    float* p   = (float*)d_ws;
    float* qb  = p; p += SZ_QK;
    float* kb  = p; p += SZ_QK;
    float* vb  = p; p += SZ_V;
    float* vl  = p; p += SZ_V;
    float* ob  = p; p += SZ_V;
    float* yb  = p; p += SZ_Y;
    float* hb  = p; p += SZ_H;
    float* h2b = p; p += SZ_H;

    dim3 blk(256);

    // q, k, v projections (WMMA GEMMs)
    gemm_wmma_kernel<<<dim3(BSZ, NHKD  / 128), blk, 0, stream>>>(q_w, x, q_b, nullptr, nullptr, qb, NHKD,  BDIM, 0);
    gemm_wmma_kernel<<<dim3(BSZ, NHKD  / 128), blk, 0, stream>>>(k_w, x, k_b, nullptr, nullptr, kb, NHKD,  BDIM, 0);
    gemm_wmma_kernel<<<dim3(BSZ, DHTOT / 128), blk, 0, stream>>>(v_w, x, v_b, nullptr, nullptr, vb, DHTOT, BDIM, 0);

    // depthwise local-V branch
    {
        int total = BSZ * DHTOT * NTOK;
        dwconv3x3_kernel<<<dim3((total + 255) / 256), blk, 0, stream>>>(vb, lv_w, lv_b, vl, DHTOT, 0, total);
    }

    // fused attention -> o = relu(attn@v^T + v_local)
    attn_kernel<<<dim3(BSZ), blk, 0, stream>>>(qb, kb, vb, vl, th1_w, th1_b, th2_w, th2_b,
                                               attn_bias, bias_idxs, ob);

    // proj + residual: y = x + ls1 * (proj(o) + b)
    gemm_wmma_kernel<<<dim3(BSZ, BDIM / 128), blk, 0, stream>>>(proj_w, ob, proj_b, x, ls1, yb, BDIM, DHTOT, 2);

    // MLP: fc1+relu -> dw3x3+relu -> fc2 + residual into d_out
    gemm_wmma_kernel<<<dim3(BSZ, HIDDEN / 128), blk, 0, stream>>>(fc1_w, yb, fc1_b, nullptr, nullptr, hb, HIDDEN, BDIM, 1);
    {
        int total = BSZ * HIDDEN * NTOK;
        dwconv3x3_kernel<<<dim3((total + 255) / 256), blk, 0, stream>>>(hb, mid_w, mid_b, h2b, HIDDEN, 1, total);
    }
    gemm_wmma_kernel<<<dim3(BSZ, BDIM / 128), blk, 0, stream>>>(fc2_w, h2b, fc2_b, yb, ls2, (float*)d_out, BDIM, HIDDEN, 2);
}
